// HardDictRepresentation_13950053777486
// MI455X (gfx1250) — matrix operand order repaired and checked by reading, not compiled, not run on hardware
//
#include <hip/hip_runtime.h>
#include <hip/hip_bf16.h>

// ---------------------------------------------------------------------------
// Problem: B=2, S=2048 -> T=4096 tokens, D=1024, DICT=16384, TOP_K=8.
// out = (reconstructed [T,D], sparse_coeffs [T,DICT]) fp32, concatenated.
// ---------------------------------------------------------------------------
#define T_TOK   4096
#define D_DIM   1024
#define DICT_N  16384
#define TOPK    8

// GEMM tiling: workgroup = 64 tokens x 2048 dict cols (1 of 8 N-partitions).
#define BM        64                 // tokens per workgroup
#define NSPLIT    8                  // dict partitions (grid.y)
#define NCOLS     (DICT_N / NSPLIT)  // 2048 columns per WG
#define BN        256                // columns per N-block (8 waves x 32)
#define KT        32                 // K per WMMA step (bf16 16x16x32)
#define NB_ITERS  (NCOLS / BN)       // 8
#define KK_ITERS  (D_DIM / KT)       // 32
#define TG        (T_TOK / BM)       // 64 token groups (grid.x)

typedef __attribute__((ext_vector_type(16))) __bf16 v16bf;
typedef __attribute__((ext_vector_type(8)))  float  v8f;
typedef unsigned int u32x4 __attribute__((ext_vector_type(4)));
typedef int          i32x4 __attribute__((ext_vector_type(4)));
typedef int          i32x8 __attribute__((ext_vector_type(8)));

struct U128x2 { uint4 lo, hi; };     // 32 bytes == one v16bf fragment

#if defined(__has_builtin)
#  if __has_builtin(__builtin_amdgcn_tensor_load_to_lds)
#    define HAS_TDM 1
#  endif
#endif
#ifndef HAS_TDM
#  define HAS_TDM 0
#endif

__device__ __forceinline__ unsigned short f2bf(float f) {
    unsigned int u = __builtin_bit_cast(unsigned int, f);
    unsigned int r = u + 0x7FFFu + ((u >> 16) & 1u);
    return (unsigned short)(r >> 16);
}

#if HAS_TDM
// 2-D TDM: tile_rows rows of tile_k 2-byte elements, global row stride
// row_stride elements, packed contiguously into LDS at lds_byte_off.
// clang-23 builtin arity: (u32x4 g0, i32x8 g1, i32x4 g2, i32x4 g3, i32x8, cpol)
__device__ __forceinline__ void tdm_load_2d(unsigned lds_byte_off,
                                            const void* gaddr,
                                            unsigned tile_k,
                                            unsigned tile_rows,
                                            unsigned row_stride) {
    const unsigned long long ga = (unsigned long long)(size_t)gaddr;
    u32x4 g0;
    g0.x = 1u;                                             // count=1 (valid)
    g0.y = lds_byte_off;                                   // lds_addr (bytes)
    g0.z = (unsigned)ga;                                   // global_addr lo
    g0.w = (unsigned)((ga >> 32) & 0x01FFFFFFu) | (2u << 30); // hi | type=2
    i32x8 g1;
    g1[0] = (int)(1u << 16);                   // data_size=2B, mask=0, no pad
    g1[1] = (int)(0xFFFFu << 16);              // tensor_dim0[15:0]  (huge)
    g1[2] = (int)(0x7FFFu | (0xFFFFu << 16));  // td0[31:16] | td1[15:0]
    g1[3] = (int)(0x7FFFu | (tile_k << 16));   // td1[31:16] | tile_dim0
    g1[4] = (int)(tile_rows & 0xFFFFu);        // tile_dim1 (tile_dim2=0)
    g1[5] = (int)row_stride;                   // tensor_dim0_stride[31:0]
    g1[6] = 0;
    g1[7] = 0;
    const i32x4 z4 = {0, 0, 0, 0};
    const i32x8 z8 = {0, 0, 0, 0, 0, 0, 0, 0};
    __builtin_amdgcn_tensor_load_to_lds(g0, g1, z4, z4, z8, 0);
}
#endif

// ---------------------------------------------------------------------------
// Kernel 1: LayerNorm(x) -> bf16, one block per token.
// ---------------------------------------------------------------------------
__global__ __launch_bounds__(256) void k_ln_bf16(
    const float* __restrict__ x, const float* __restrict__ gamma,
    const float* __restrict__ beta, unsigned short* __restrict__ out) {
    __shared__ float rs[256], rs2[256];
    const int tok = blockIdx.x;
    const int tid = threadIdx.x;
    const int dim = tid * 4;

    const float4 xv = *(const float4*)&x[(size_t)tok * D_DIM + dim];
    rs[tid]  = xv.x + xv.y + xv.z + xv.w;
    rs2[tid] = xv.x * xv.x + xv.y * xv.y + xv.z * xv.z + xv.w * xv.w;
    __syncthreads();
    for (int off = 128; off > 0; off >>= 1) {
        if (tid < off) { rs[tid] += rs[tid + off]; rs2[tid] += rs2[tid + off]; }
        __syncthreads();
    }
    const float mu   = rs[0] * (1.0f / D_DIM);
    const float var  = rs2[0] * (1.0f / D_DIM) - mu * mu;
    const float rsig = rsqrtf(var + 1e-5f);

    const float4 g4 = *(const float4*)&gamma[dim];
    const float4 b4 = *(const float4*)&beta[dim];
    union { unsigned short us[4]; uint2 u; } pk;
    pk.us[0] = f2bf((xv.x - mu) * rsig * g4.x + b4.x);
    pk.us[1] = f2bf((xv.y - mu) * rsig * g4.y + b4.y);
    pk.us[2] = f2bf((xv.z - mu) * rsig * g4.z + b4.z);
    pk.us[3] = f2bf((xv.w - mu) * rsig * g4.w + b4.w);
    *(uint2*)&out[(size_t)tok * D_DIM + dim] = pk.u;
}

// ---------------------------------------------------------------------------
// Kernel 2: W_enc fp32 -> bf16.
// ---------------------------------------------------------------------------
__global__ __launch_bounds__(256) void k_w_bf16(
    const float* __restrict__ w, unsigned short* __restrict__ out, int n4) {
    const int i = blockIdx.x * blockDim.x + threadIdx.x;
    if (i >= n4) return;
    const float4 v = *(const float4*)&w[(size_t)i * 4];
    union { unsigned short us[4]; uint2 u; } pk;
    pk.us[0] = f2bf(v.x); pk.us[1] = f2bf(v.y);
    pk.us[2] = f2bf(v.z); pk.us[3] = f2bf(v.w);
    *(uint2*)&out[(size_t)i * 4] = pk.u;
}

// ---------------------------------------------------------------------------
// Kernel 3: WMMA GEMM (logits) + streaming partial top-8 + zero-fill.
// grid = (TG, NSPLIT). WG owns 64 tokens x 2048 cols. A-tile (128 KB bf16)
// resident in LDS; B tiles (256x32 bf16) double-buffered via TDM, next tile
// DMA'd while current is consumed. Wave w covers cols [w*32, w*32+32):
// 4 M-tiles x 2 N-tiles = 8 v_wmma per K-step, 64 acc VGPRs.
// ---------------------------------------------------------------------------
__global__ __launch_bounds__(256) void k_gemm_topk(
    const unsigned short* __restrict__ normed_bf,
    const unsigned short* __restrict__ Wenc_bf,
    float* __restrict__ sparse_out,
    float* __restrict__ cand_v, int* __restrict__ cand_i) {
    __shared__ __align__(16) unsigned short sA[BM * D_DIM];   // 128 KB
    __shared__ __align__(16) unsigned char  sU[32768];        // 32 KB union

    unsigned short* sB   = (unsigned short*)sU;  // 2 x 16 KB (double buffer)
    float*          sLog = (float*)sU;           // 32 x 256 f32 (half spill)

    const int tid  = threadIdx.x;
    const int wave = tid >> 5;
    const int lane = tid & 31;
    const int tok0 = blockIdx.x * BM;
    const int nsb  = blockIdx.y * NCOLS;
    const int mrow = lane & 15;
    const int hi16 = lane >> 4;
    const int uw   = __builtin_amdgcn_readfirstlane(wave);
#if HAS_TDM
    const unsigned sA_off = (unsigned)(size_t)(void*)sA;
    const unsigned sU_off = (unsigned)(size_t)(void*)sU;
#endif

    // ---- stage resident A tile: 64 token rows x 1024 bf16 (128 KB) ----
#if HAS_TDM
    if (uw == 0) {
        tdm_load_2d(sA_off, normed_bf + (size_t)tok0 * D_DIM, D_DIM, BM, D_DIM);
        __builtin_amdgcn_s_wait_tensorcnt(0);
    }
#else
    {
        const uint4* src = (const uint4*)(normed_bf + (size_t)tok0 * D_DIM);
        uint4* dst = (uint4*)sA;
        for (int i = tid; i < (BM * D_DIM) / 8; i += 256) dst[i] = src[i];
    }
#endif
    __syncthreads();

    // per-thread running top-8: half H covers token H*32 + (tid&31),
    // column stripe chunk = tid>>5 (32 cols of each 256-col block).
    float bv[2][TOPK]; int bi[2][TOPK];
#pragma unroll
    for (int h = 0; h < 2; ++h)
#pragma unroll
        for (int j = 0; j < TOPK; ++j) { bv[h][j] = -3.402823466e38f; bi[h][j] = 0; }

    for (int nb = 0; nb < NB_ITERS; ++nb) {
        const int n0g = nsb + nb * BN;      // global dict column base

        // ---- preload B buffer 0 for k=0 ----
#if HAS_TDM
        if (uw == 0) {
            tdm_load_2d(sU_off, Wenc_bf + (size_t)n0g * D_DIM, KT, BN, D_DIM);
            __builtin_amdgcn_s_wait_tensorcnt(0);
        }
#else
        {
            const uint4* s = (const uint4*)(Wenc_bf + (size_t)(n0g + tid) * D_DIM);
            uint4* d = (uint4*)(sB + tid * KT);
            d[0] = s[0]; d[1] = s[1]; d[2] = s[2]; d[3] = s[3];
        }
#endif
        __syncthreads();

        v8f acc[4][2];
#pragma unroll
        for (int Mt = 0; Mt < 4; ++Mt)
#pragma unroll
            for (int Nt = 0; Nt < 2; ++Nt) { v8f z = {}; acc[Mt][Nt] = z; }

        for (int kk = 0; kk < KK_ITERS; ++kk) {
            const int cur = kk & 1;
            // ---- DMA next K-tile into the other buffer (overlaps compute) ----
            if (kk + 1 < KK_ITERS) {
#if HAS_TDM
                if (uw == 0)
                    tdm_load_2d(sU_off + (cur ^ 1) * 16384,
                                Wenc_bf + (size_t)n0g * D_DIM + (kk + 1) * KT,
                                KT, BN, D_DIM);
#else
                const uint4* s = (const uint4*)(Wenc_bf +
                    (size_t)(n0g + tid) * D_DIM + (kk + 1) * KT);
                uint4* d = (uint4*)(sB + (cur ^ 1) * (BN * KT) + tid * KT);
                d[0] = s[0]; d[1] = s[1]; d[2] = s[2]; d[3] = s[3];
#endif
            }
            const unsigned short* sBc = sB + cur * (BN * KT);
            const int k0 = kk * KT;

            // A fragments: lanes0-15 K+0..7 / K+16..23, lanes16-31 +8
            v16bf af[4];
#pragma unroll
            for (int Mt = 0; Mt < 4; ++Mt) {
                const unsigned short* ap =
                    sA + (Mt * 16 + mrow) * D_DIM + k0 + hi16 * 8;
                U128x2 u;
                u.lo = *(const uint4*)ap;
                u.hi = *(const uint4*)(ap + 16);
                af[Mt] = __builtin_bit_cast(v16bf, u);
            }
#pragma unroll
            for (int Nt = 0; Nt < 2; ++Nt) {
                const int ncol = wave * 32 + Nt * 16 + mrow;
                const unsigned short* bp = sBc + ncol * KT + hi16 * 8;
                U128x2 u;
                u.lo = *(const uint4*)bp;
                u.hi = *(const uint4*)(bp + 16);
                const v16bf bf = __builtin_bit_cast(v16bf, u);
#pragma unroll
                for (int Mt = 0; Mt < 4; ++Mt)
                    acc[Mt][Nt] = __builtin_amdgcn_wmma_f32_16x16x32_bf16(
                        false, af[Mt], false, bf, (short)0, acc[Mt][Nt],
                        false, false);
            }
#if HAS_TDM
            if (uw == 0) __builtin_amdgcn_s_wait_tensorcnt(0);
#endif
            __syncthreads();
        }

        // ---- zero-fill this block's region of the dense sparse output ----
        {
            const float4 z = make_float4(0.f, 0.f, 0.f, 0.f);
#pragma unroll
            for (int q = 0; q < 16; ++q) {
                const int flat = q * 1024 + tid * 4;   // 0..16383, coalesced
                const int m = flat >> 8;
                const int c = flat & 255;
                *(float4*)&sparse_out[(size_t)(tok0 + m) * DICT_N + n0g + c] = z;
            }
        }

        // ---- spill + scan logits in two 32-token halves (32 KB each) ----
#pragma unroll
        for (int H = 0; H < 2; ++H) {
            // spill acc tiles Mt in {2H, 2H+1}; C layout: M=r+8*hi16, N=lane&15
#pragma unroll
            for (int MtH = 0; MtH < 2; ++MtH) {
#pragma unroll
                for (int Nt = 0; Nt < 2; ++Nt) {
                    const int ncol = wave * 32 + Nt * 16 + mrow;
                    const v8f c = acc[H * 2 + MtH][Nt];
#pragma unroll
                    for (int r = 0; r < 8; ++r)
                        sLog[(MtH * 16 + r + hi16 * 8) * BN + ncol] = c[r];
                }
            }
            __syncthreads();
            {
                const int m32   = tid & 31;
                const int chunk = tid >> 5;
                const float* row = sLog + m32 * BN + chunk * 32;
                for (int j = 0; j < 32; ++j) {
                    float v = row[j];
                    if (v > bv[H][TOPK - 1]) {
                        int n = n0g + chunk * 32 + j;
#pragma unroll
                        for (int q = 0; q < TOPK; ++q) {
                            const bool gt = v > bv[H][q];
                            const float tv = gt ? bv[H][q] : v;
                            bv[H][q] = gt ? v : bv[H][q]; v = tv;
                            const int tn = gt ? bi[H][q] : n;
                            bi[H][q] = gt ? n : bi[H][q]; n = tn;
                        }
                    }
                }
            }
            __syncthreads();
        }
    }

    // ---- WG-level merge: 8 chunk-lists of 8 -> top-8 per token ----
    float* cvv = (float*)sU;                 // 64 tok x 8 chunks x 8 (16 KB)
    int*   cii = (int*)(sU + 16384);         // 16 KB
    {
        const int m32 = tid & 31, chunk = tid >> 5;
#pragma unroll
        for (int H = 0; H < 2; ++H) {
            const int base = ((H * 32 + m32) * 8 + chunk) * TOPK;
#pragma unroll
            for (int j = 0; j < TOPK; ++j) {
                cvv[base + j] = bv[H][j];
                cii[base + j] = bi[H][j];
            }
        }
    }
    __syncthreads();
    if (tid < BM) {
        float fv[TOPK]; int fi[TOPK];
#pragma unroll
        for (int j = 0; j < TOPK; ++j) { fv[j] = -3.402823466e38f; fi[j] = 0; }
        for (int c = 0; c < 8 * TOPK; ++c) {
            float v = cvv[tid * 8 * TOPK + c];
            int   n = cii[tid * 8 * TOPK + c];
            if (v > fv[TOPK - 1]) {
#pragma unroll
                for (int q = 0; q < TOPK; ++q) {
                    const bool gt = v > fv[q];
                    const float tv = gt ? fv[q] : v; fv[q] = gt ? v : fv[q]; v = tv;
                    const int   tn = gt ? fi[q] : n; fi[q] = gt ? n : fi[q]; n = tn;
                }
            }
        }
        const size_t base = ((size_t)(tok0 + tid) * NSPLIT + blockIdx.y) * TOPK;
#pragma unroll
        for (int j = 0; j < TOPK; ++j) {
            cand_v[base + j] = fv[j];
            cand_i[base + j] = fi[j];
        }
    }
}

// ---------------------------------------------------------------------------
// Kernel 4: merge the 8 partition candidate lists (64 entries) -> final
// top-8, scatter into the zeroed dense output, reconstruct in fp32.
// ---------------------------------------------------------------------------
__global__ __launch_bounds__(256) void k_merge_recon(
    const float* __restrict__ W_dict,
    const float* __restrict__ cand_v, const int* __restrict__ cand_i,
    float* __restrict__ recon, float* __restrict__ sparse_out) {
    const int tok = blockIdx.x;
    const int tid = threadIdx.x;
    __shared__ float svv[64]; __shared__ int sii[64];
    __shared__ float fvs[TOPK]; __shared__ int fis[TOPK];
    if (tid < 64) {
        svv[tid] = cand_v[(size_t)tok * 64 + tid];
        sii[tid] = cand_i[(size_t)tok * 64 + tid];
    }
    __syncthreads();
    if (tid == 0) {
        float fv[TOPK]; int fi[TOPK];
#pragma unroll
        for (int j = 0; j < TOPK; ++j) { fv[j] = -3.402823466e38f; fi[j] = 0; }
        for (int c = 0; c < 64; ++c) {
            float v = svv[c]; int n = sii[c];
            if (v > fv[TOPK - 1]) {
#pragma unroll
                for (int q = 0; q < TOPK; ++q) {
                    const bool gt = v > fv[q];
                    const float tv = gt ? fv[q] : v; fv[q] = gt ? v : fv[q]; v = tv;
                    const int   tn = gt ? fi[q] : n; fi[q] = gt ? n : fi[q]; n = tn;
                }
            }
        }
#pragma unroll
        for (int j = 0; j < TOPK; ++j) { fvs[j] = fv[j]; fis[j] = fi[j]; }
    }
    __syncthreads();
    if (tid < TOPK)
        sparse_out[(size_t)tok * DICT_N + fis[tid]] = fvs[tid];

    const int dim = tid * 4;
    float4 a = make_float4(0.f, 0.f, 0.f, 0.f);
#pragma unroll
    for (int j = 0; j < TOPK; ++j) {
        const float4 w = *(const float4*)&W_dict[(size_t)fis[j] * D_DIM + dim];
        const float v = fvs[j];
        a.x += v * w.x; a.y += v * w.y; a.z += v * w.z; a.w += v * w.w;
    }
    *(float4*)&recon[(size_t)tok * D_DIM + dim] = a;
}

// ---------------------------------------------------------------------------
extern "C" void kernel_launch(void* const* d_in, const int* in_sizes, int n_in,
                              void* d_out, int out_size, void* d_ws, size_t ws_size,
                              hipStream_t stream) {
    const float* x      = (const float*)d_in[0];
    const float* gamma  = (const float*)d_in[1];
    const float* beta   = (const float*)d_in[2];
    const float* W_enc  = (const float*)d_in[3];
    const float* W_dict = (const float*)d_in[4];

    float* recon  = (float*)d_out;                       // [T, D]
    float* sparse = recon + (size_t)T_TOK * D_DIM;       // [T, DICT]

    // ws: normed bf16 (8MB) | W_enc bf16 (32MB) | cand vals (1MB) | cand idx
    unsigned short* normed_bf = (unsigned short*)d_ws;
    unsigned short* Wenc_bf   = normed_bf + (size_t)T_TOK * D_DIM;
    float* cand_v = (float*)(Wenc_bf + (size_t)DICT_N * D_DIM);
    int*   cand_i = (int*)(cand_v + (size_t)T_TOK * NSPLIT * TOPK);

    k_ln_bf16<<<T_TOK, 256, 0, stream>>>(x, gamma, beta, normed_bf);
    k_w_bf16<<<(DICT_N * D_DIM / 4 + 255) / 256, 256, 0, stream>>>(
        W_enc, Wenc_bf, DICT_N * D_DIM / 4);
    k_gemm_topk<<<dim3(TG, NSPLIT), 256, 0, stream>>>(
        normed_bf, Wenc_bf, sparse, cand_v, cand_i);
    k_merge_recon<<<T_TOK, 256, 0, stream>>>(
        W_dict, cand_v, cand_i, recon, sparse);
}
